// GPTLanguageModel_36962488549944
// MI455X (gfx1250) — matrix operand order repaired and checked by reading, not compiled, not run on hardware
//
#include <hip/hip_runtime.h>
#include <hip/hip_bf16.h>

// ---------------- types / helpers ----------------
typedef unsigned short u16;
typedef __attribute__((ext_vector_type(16))) __bf16 v16bf;
typedef __attribute__((ext_vector_type(8)))  float  v8f;
typedef __attribute__((ext_vector_type(4)))  unsigned int u32x4;

__device__ __forceinline__ u16 to_bf16(float f) {
    unsigned int u = __float_as_uint(f);
    u += 0x7FFFu + ((u >> 16) & 1u);          // round-to-nearest-even
    return (u16)(u >> 16);
}
__device__ __forceinline__ float from_bf16(u16 h) {
    return __uint_as_float(((unsigned int)h) << 16);
}

// Model dims
constexpr int V_  = 50262;
constexpr int Vp_ = 50304;   // ceil(V/64)*64, zero-padded rows in WoutT
constexpr int E_  = 256;
constexpr int H_  = 4;
constexpr int D_  = 64;
constexpr int L_  = 4;
constexpr int T_  = 768;
constexpr int B_  = 8;
constexpr int FF_ = 1024;
constexpr int M_  = B_ * T_; // 6144

// ---------------- generic batched WMMA GEMM ----------------
// C[m,n] = sum_k A_bf16[m,k] * B_bf16[k,n]  with Bt = B^T stored [N,K].
// Batch index z -> (b = z/nH, h = z%nH); per-operand (b,h) strides in elements.
// Epilogue: + bias[n], + Res (same layout as C), ReLU, write f32 and/or bf16.
// Block: 256 threads = 8 waves stacked along M; per-wave tile = 16M x 64N
// (4 accumulators, A-fragment reused across 4 WMMAs -> ~4x arithmetic
// intensity vs one-tile-per-wave). Block tile = 128M x 64N.
// Requires: N tile extent (64) fully in-bounds for B loads (all N here are
// multiples of 64; Wout is padded to Vp_ rows). Store guarded by col < N.
__global__ __launch_bounds__(256) void wmma_gemm_kernel(
    const u16* __restrict__ A,  long long As_b, long long As_h, int lda,
    const u16* __restrict__ Bt, long long Bs_b, long long Bs_h, int ldb,
    const float* __restrict__ bias,
    const float* __restrict__ Res,
    float* __restrict__ Cf, u16* __restrict__ Cb,
    long long Cs_b, long long Cs_h, int ldc,
    int M, int N, int K, int nH, int relu)
{
    const int z  = blockIdx.z;
    const int bb = z / nH, hh = z % nH;
    A  += (long long)bb * As_b + (long long)hh * As_h;
    Bt += (long long)bb * Bs_b + (long long)hh * Bs_h;
    const long long coff = (long long)bb * Cs_b + (long long)hh * Cs_h;

    const int wave = threadIdx.x >> 5;
    const int lane = threadIdx.x & 31;
    const int mBase = blockIdx.y * 128 + wave * 16;
    const int nBase = blockIdx.x * 64;
    const int lhalf = lane >> 4;      // 0: lanes 0-15, 1: lanes 16-31
    const int l16   = lane & 15;

    // A frag (16x32 bf16): lane row = l16; chunks K = lhalf*8 + [0..7] and +16.
    const u16* Ap = A  + (long long)(mBase + l16) * lda + (lhalf << 3);
    // B frags from Bt[N,K]: lane col = l16 (+16*j); K = lhalf*8 + [0..7] and +16.
    const u16* Bp = Bt + (long long)(nBase + l16) * ldb + (lhalf << 3);
    const long long bstep = (long long)16 * ldb;

    v8f acc[4] = {};
    for (int k = 0; k < K; k += 32) {
        v16bf a;
        ((u32x4*)&a)[0] = *(const u32x4*)(Ap);
        ((u32x4*)&a)[1] = *(const u32x4*)(Ap + 16);
        v16bf bf[4];
#pragma unroll
        for (int j = 0; j < 4; ++j) {
            const u16* bp = Bp + j * bstep;
            ((u32x4*)&bf[j])[0] = *(const u32x4*)(bp);
            ((u32x4*)&bf[j])[1] = *(const u32x4*)(bp + 16);
        }
        __builtin_prefetch(Ap + 32, 0, 1);   // global_prefetch_b8 next K-chunk
        __builtin_prefetch(Bp + 32, 0, 1);
#pragma unroll
        for (int j = 0; j < 4; ++j)
            acc[j] = __builtin_amdgcn_wmma_f32_16x16x32_bf16(
                         false, a, false, bf[j], (short)0, acc[j], false, false);
        Ap += 32; Bp += 32;
    }

    // C layout per 16x16 tile: VGPR r -> row = r + 8*lhalf, col = l16.
    const int rbase = mBase + (lhalf << 3);
#pragma unroll
    for (int j = 0; j < 4; ++j) {
        const int col  = nBase + j * 16 + l16;
        const bool cok = col < N;
        const float bv = (bias && cok) ? bias[col] : 0.0f;
        for (int r = 0; r < 8; ++r) {
            const long long off = coff + (long long)(rbase + r) * ldc + col;
            float v = acc[j][r] + bv;
            if (Res && cok) v += Res[off];
            if (relu) v = fmaxf(v, 0.0f);
            if (cok) {
                if (Cf) Cf[off] = v;
                if (Cb) Cb[off] = to_bf16(v);
            }
        }
    }
}

// ---------------- weight transpose+convert: [z][K][N] f32 -> [z][Npad][K] bf16 ----------------
__global__ __launch_bounds__(256) void wtrans_kernel(
    const float* __restrict__ W, u16* __restrict__ Wt, int K, int N, int Npad)
{
    const long long tot = (long long)Npad * K;
    const long long idx = (long long)blockIdx.x * 256 + threadIdx.x;
    if (idx >= tot) return;
    const long long z = blockIdx.y;
    const int n = (int)(idx / K), k = (int)(idx % K);
    const float v = (n < N) ? W[z * (long long)K * N + (long long)k * N + n] : 0.0f;
    Wt[z * tot + idx] = to_bf16(v);
}

// Wq/Wk/Wv [L,H,E,D] f32 -> Wt [L, H*D, E] bf16  (rows n = h*D+d, cols k = e)
__global__ __launch_bounds__(256) void qkvtrans_kernel(
    const float* __restrict__ W, u16* __restrict__ Wt)
{
    const int idx = blockIdx.x * 256 + threadIdx.x;   // L*H*E*D = 262144 exact
    const int d = idx % D_;
    const int e = (idx / D_) % E_;
    const int h = (idx / (D_ * E_)) % H_;
    const int l = idx / (D_ * E_ * H_);
    Wt[(((long long)l * E_ + h * D_ + d)) * E_ + e] = to_bf16(W[idx]);
}

// v bf16 [B*T, E] (head cols) -> vT bf16 [B*H, D, T]
__global__ __launch_bounds__(256) void vtrans_kernel(
    const u16* __restrict__ v, u16* __restrict__ vT)
{
    const int z = blockIdx.y;          // b*H + h
    const int b = z / H_, h = z % H_;
    const int idx = blockIdx.x * 256 + threadIdx.x;   // T*D = 49152 exact
    const int t = idx / D_, d = idx % D_;
    vT[(long long)z * D_ * T_ + (long long)d * T_ + t] =
        v[(long long)b * T_ * E_ + (long long)t * E_ + h * D_ + d];
}

// ---------------- embedding: x = tok_emb[inputs] + pos_emb ----------------
__global__ __launch_bounds__(256) void embed_kernel(
    const int* __restrict__ inputs, const float* __restrict__ tok,
    const float* __restrict__ pos, float* __restrict__ x)
{
    const long long idx = (long long)blockIdx.x * 256 + threadIdx.x; // M_*E_ exact
    const int e = (int)(idx & (E_ - 1));
    const int m = (int)(idx >> 8);
    const int t = m % T_;
    const int tk = inputs[m];
    x[idx] = tok[(long long)tk * E_ + e] + pos[(long long)t * E_ + e];
}

// ---------------- layernorm (row=256) -> bf16 ----------------
__global__ __launch_bounds__(256) void ln_kernel(
    const float* __restrict__ x, const float* __restrict__ g,
    const float* __restrict__ b, u16* __restrict__ out)
{
    __shared__ float s1[256], s2[256];
    const int row = blockIdx.x, e = threadIdx.x;
    const float v = x[(long long)row * E_ + e];
    s1[e] = v; s2[e] = v * v;
    __syncthreads();
    for (int st = 128; st > 0; st >>= 1) {
        if (e < st) { s1[e] += s1[e + st]; s2[e] += s2[e + st]; }
        __syncthreads();
    }
    const float mu  = s1[0] * (1.0f / E_);
    const float var = s2[0] * (1.0f / E_) - mu * mu;
    const float inv = rsqrtf(var + 1e-5f);
    out[(long long)row * E_ + e] = to_bf16((v - mu) * inv * g[e] + b[e]);
}

// ---------------- causal softmax over scores (bf16 in-place) ----------------
__global__ __launch_bounds__(256) void attn_softmax_kernel(
    u16* __restrict__ sc, float scale)
{
    __shared__ float red[256];
    const int r = blockIdx.x;          // b*H*T + h*T + t
    const int t = r % T_;
    u16* row = sc + (long long)r * T_;
    const int tid = threadIdx.x;

    float mx = -3.0e38f;
    for (int s = tid; s < T_; s += 256)
        if (s <= t) mx = fmaxf(mx, from_bf16(row[s]) * scale);
    red[tid] = mx; __syncthreads();
    for (int st = 128; st > 0; st >>= 1) {
        if (tid < st) red[tid] = fmaxf(red[tid], red[tid + st]);
        __syncthreads();
    }
    mx = red[0]; __syncthreads();

    float sum = 0.0f;
    for (int s = tid; s < T_; s += 256)
        if (s <= t) sum += __expf(from_bf16(row[s]) * scale - mx);
    red[tid] = sum; __syncthreads();
    for (int st = 128; st > 0; st >>= 1) {
        if (tid < st) red[tid] += red[tid + st];
        __syncthreads();
    }
    const float inv = 1.0f / red[0];
    for (int s = tid; s < T_; s += 256) {
        const float p = (s <= t) ? __expf(from_bf16(row[s]) * scale - mx) * inv : 0.0f;
        row[s] = to_bf16(p);
    }
}

// ---------------- cross-entropy: per-row lse - logit[target] ----------------
__global__ __launch_bounds__(256) void ce_rows_kernel(
    const float* __restrict__ logits, const int* __restrict__ targets,
    float* __restrict__ rowloss)
{
    __shared__ float red[256];
    const int r = blockIdx.x, tid = threadIdx.x;
    const float* row = logits + (long long)r * V_;
    float mx = -3.0e38f;
    for (int s = tid; s < V_; s += 256) mx = fmaxf(mx, row[s]);
    red[tid] = mx; __syncthreads();
    for (int st = 128; st > 0; st >>= 1) {
        if (tid < st) red[tid] = fmaxf(red[tid], red[tid + st]);
        __syncthreads();
    }
    mx = red[0]; __syncthreads();
    float sum = 0.0f;
    for (int s = tid; s < V_; s += 256) sum += __expf(row[s] - mx);
    red[tid] = sum; __syncthreads();
    for (int st = 128; st > 0; st >>= 1) {
        if (tid < st) red[tid] += red[tid + st];
        __syncthreads();
    }
    if (tid == 0) rowloss[r] = (mx + __logf(red[0])) - row[targets[r]];
}

__global__ __launch_bounds__(256) void ce_reduce_kernel(
    const float* __restrict__ rowloss, float* __restrict__ out)
{
    __shared__ float red[256];
    const int tid = threadIdx.x;
    float s = 0.0f;
    for (int i = tid; i < M_; i += 256) s += rowloss[i];
    red[tid] = s; __syncthreads();
    for (int st = 128; st > 0; st >>= 1) {
        if (tid < st) red[tid] += red[tid + st];
        __syncthreads();
    }
    if (tid == 0) out[0] = red[0] / (float)M_;
}

// ---------------- host launcher ----------------
extern "C" void kernel_launch(void* const* d_in, const int* in_sizes, int n_in,
                              void* d_out, int out_size, void* d_ws, size_t ws_size,
                              hipStream_t stream)
{
    (void)in_sizes; (void)n_in; (void)out_size; (void)ws_size;
    const int*   inputs = (const int*)  d_in[0];
    const int*   targets= (const int*)  d_in[1];
    const float* tok    = (const float*)d_in[2];
    const float* pos    = (const float*)d_in[3];
    const float* Wq     = (const float*)d_in[4];
    const float* Wk     = (const float*)d_in[5];
    const float* Wv     = (const float*)d_in[6];
    const float* Wo     = (const float*)d_in[7];
    const float* bo     = (const float*)d_in[8];
    const float* ln1g   = (const float*)d_in[9];
    const float* ln1b   = (const float*)d_in[10];
    const float* ln2g   = (const float*)d_in[11];
    const float* ln2b   = (const float*)d_in[12];
    const float* W1     = (const float*)d_in[13];
    const float* b1     = (const float*)d_in[14];
    const float* W2     = (const float*)d_in[15];
    const float* b2     = (const float*)d_in[16];
    const float* lnfg   = (const float*)d_in[17];
    const float* lnfb   = (const float*)d_in[18];
    const float* Wout   = (const float*)d_in[19];
    const float* bout   = (const float*)d_in[20];

    float* logits = (float*)d_out;
    float* lossp  = logits + (long long)M_ * V_;

    // workspace carve-up (256B aligned)
    char* w = (char*)d_ws;
    auto alloc = [&](size_t bytes) -> void* {
        void* p = (void*)w; w += (bytes + 255) & ~(size_t)255; return p;
    };
    float* x     = (float*)alloc((size_t)M_ * E_ * 4);
    u16*   hbf   = (u16*)  alloc((size_t)M_ * E_ * 2);
    u16*   qb    = (u16*)  alloc((size_t)M_ * E_ * 2);
    u16*   kb    = (u16*)  alloc((size_t)M_ * E_ * 2);
    u16*   vb    = (u16*)  alloc((size_t)M_ * E_ * 2);
    u16*   ob    = (u16*)  alloc((size_t)M_ * E_ * 2);
    u16*   vT    = (u16*)  alloc((size_t)B_ * H_ * D_ * T_ * 2);
    u16*   sc    = (u16*)  alloc((size_t)B_ * H_ * T_ * T_ * 2);   // scores -> att
    u16*   ffb   = (u16*)  alloc((size_t)M_ * FF_ * 2);
    u16*   WqT   = (u16*)  alloc((size_t)L_ * E_ * E_ * 2);
    u16*   WkT   = (u16*)  alloc((size_t)L_ * E_ * E_ * 2);
    u16*   WvT   = (u16*)  alloc((size_t)L_ * E_ * E_ * 2);
    u16*   WoT   = (u16*)  alloc((size_t)L_ * E_ * E_ * 2);
    u16*   W1T   = (u16*)  alloc((size_t)L_ * FF_ * E_ * 2);
    u16*   W2T   = (u16*)  alloc((size_t)L_ * E_ * FF_ * 2);
    u16*   WoutT = (u16*)  alloc((size_t)Vp_ * E_ * 2);
    float* rloss = (float*)alloc((size_t)M_ * 4);

    auto gemm = [&](const u16* A, long long Asb, long long Ash, int lda,
                    const u16* Bt, long long Bsb, long long Bsh, int ldb,
                    const float* bias, const float* Res, float* Cf, u16* Cb,
                    long long Csb, long long Csh, int ldc,
                    int M, int N, int K, int Z, int nH, int relu) {
        dim3 grid((N + 63) / 64, (M + 127) / 128, Z);
        wmma_gemm_kernel<<<grid, 256, 0, stream>>>(A, Asb, Ash, lda, Bt, Bsb, Bsh, ldb,
            bias, Res, Cf, Cb, Csb, Csh, ldc, M, N, K, nH, relu);
    };

    // --- weight conversions (once per launch) ---
    wtrans_kernel<<<dim3((E_*E_)/256, L_), 256, 0, stream>>>(Wo, WoT, E_, E_, E_);
    wtrans_kernel<<<dim3((FF_*E_)/256, L_), 256, 0, stream>>>(W1, W1T, E_, FF_, FF_);
    wtrans_kernel<<<dim3((E_*FF_)/256, L_), 256, 0, stream>>>(W2, W2T, FF_, E_, E_);
    wtrans_kernel<<<dim3((Vp_*E_)/256, 1),  256, 0, stream>>>(Wout, WoutT, E_, V_, Vp_);
    qkvtrans_kernel<<<(L_*H_*E_*D_)/256, 256, 0, stream>>>(Wq, WqT);
    qkvtrans_kernel<<<(L_*H_*E_*D_)/256, 256, 0, stream>>>(Wk, WkT);
    qkvtrans_kernel<<<(L_*H_*E_*D_)/256, 256, 0, stream>>>(Wv, WvT);

    // --- embedding ---
    embed_kernel<<<(M_*E_)/256, 256, 0, stream>>>(inputs, tok, pos, x);

    const float scale = 0.125f;  // D^-0.5
    for (int l = 0; l < L_; ++l) {
        // ln1 -> hbf
        ln_kernel<<<M_, 256, 0, stream>>>(x, ln1g + l*E_, ln1b + l*E_, hbf);
        // q,k,v = hbf @ W{q,k,v}  -> bf16 [M, E] (head-major cols)
        gemm(hbf, 0,0, E_, WqT + (long long)l*E_*E_, 0,0, E_, nullptr, nullptr,
             nullptr, qb, 0,0, E_, M_, E_, E_, 1, 1, 0);
        gemm(hbf, 0,0, E_, WkT + (long long)l*E_*E_, 0,0, E_, nullptr, nullptr,
             nullptr, kb, 0,0, E_, M_, E_, E_, 1, 1, 0);
        gemm(hbf, 0,0, E_, WvT + (long long)l*E_*E_, 0,0, E_, nullptr, nullptr,
             nullptr, vb, 0,0, E_, M_, E_, E_, 1, 1, 0);
        // vT for att @ v
        vtrans_kernel<<<dim3((T_*D_)/256, B_*H_), 256, 0, stream>>>(vb, vT);
        // scores = q @ k^T   (Bt == k directly), per (b,h)
        gemm(qb, (long long)T_*E_, D_, E_,
             kb, (long long)T_*E_, D_, E_,
             nullptr, nullptr, nullptr, sc,
             (long long)H_*T_*T_, (long long)T_*T_, T_,
             T_, T_, D_, B_*H_, H_, 0);
        // causal softmax (in place, bf16)
        attn_softmax_kernel<<<B_*H_*T_, 256, 0, stream>>>(sc, scale);
        // o = att @ v  -> bf16 [M, E] head-major cols
        gemm(sc, (long long)H_*T_*T_, (long long)T_*T_, T_,
             vT, (long long)H_*D_*T_, (long long)D_*T_, T_,
             nullptr, nullptr, nullptr, ob,
             (long long)T_*E_, D_, E_,
             T_, D_, T_, B_*H_, H_, 0);
        // x = o @ Wo + bo + x   (f32, in place)
        gemm(ob, 0,0, E_, WoT + (long long)l*E_*E_, 0,0, E_,
             bo + l*E_, x, x, nullptr, 0,0, E_, M_, E_, E_, 1, 1, 0);
        // ln2 -> hbf
        ln_kernel<<<M_, 256, 0, stream>>>(x, ln2g + l*E_, ln2b + l*E_, hbf);
        // ff = relu(hbf @ W1 + b1) -> bf16 [M, FF]
        gemm(hbf, 0,0, E_, W1T + (long long)l*FF_*E_, 0,0, E_,
             b1 + l*FF_, nullptr, nullptr, ffb, 0,0, FF_, M_, FF_, E_, 1, 1, 1);
        // x = ff @ W2 + b2 + x  (f32, in place)
        gemm(ffb, 0,0, FF_, W2T + (long long)l*E_*FF_, 0,0, FF_,
             b2 + l*E_, x, x, nullptr, 0,0, E_, M_, E_, FF_, 1, 1, 0);
    }

    // final LN -> hbf
    ln_kernel<<<M_, 256, 0, stream>>>(x, lnfg, lnfb, hbf);
    // logits = hbf @ Wout + bout  (f32 straight into d_out, ldc = V, guarded N)
    gemm(hbf, 0,0, E_, WoutT, 0,0, E_,
         bout, nullptr, logits, nullptr, 0,0, V_, M_, V_, E_, 1, 1, 0);

    // cross-entropy loss
    ce_rows_kernel<<<M_, 256, 0, stream>>>(logits, targets, rloss);
    ce_reduce_kernel<<<1, 256, 0, stream>>>(rloss, lossp);
}
